// one_batch_net_70763881169472
// MI455X (gfx1250) — matrix operand order repaired and checked by reading, not compiled, not run on hardware
//
#include <hip/hip_runtime.h>
#include <hip/hip_fp16.h>

typedef __attribute__((ext_vector_type(16))) _Float16 v16h;
typedef __attribute__((ext_vector_type(8)))  float    v8f;
typedef __attribute__((ext_vector_type(4)))  unsigned int u32x4;

union FragU { v16h v; u32x4 q[2]; };

// Branch-free tanh: one v_exp_f32 + one v_rcp_f32; clamp avoids inf/inf.
__device__ __forceinline__ float fast_tanh(float x) {
  float xc = fminf(fmaxf(x, -12.f), 12.f);
  float e = __expf(2.f * xc);
  float r = __builtin_amdgcn_rcpf(e + 1.f);
  return fmaf(-2.f, r, 1.f);           // (e-1)/(e+1)
}

__device__ __forceinline__ void frag_load(FragU& a, FragU& b,
                                          const _Float16* ap, const _Float16* bp) {
  a.q[0] = *(const u32x4*)(ap);        // K = g*8 .. +7
  a.q[1] = *(const u32x4*)(ap + 16);   // K = 16+g*8 .. +7
  b.q[0] = *(const u32x4*)(bp);        // K = g*8 .. +7
  b.q[1] = *(const u32x4*)(bp + 16);   // K = 16+g*8 .. +7
}

// One 16x16 C-tile, A(MxK) row-major (LDS), B packed as N-major rows of K (global f16).
// kTiles = K/32; fragment loads are software-pipelined one K-step ahead.
__device__ __forceinline__ v8f wmma_tile(const _Float16* A, int lda,
                                         const _Float16* B, int ldb,
                                         int kTiles, int lane, v8f acc)
{
  const int m = lane & 15;       // A row / B column / C column
  const int g = lane >> 4;       // lane-group selects K half
  const _Float16* ap = A + m * lda + g * 8;
  const _Float16* bp = B + m * ldb + g * 8;
  FragU a, b;
  frag_load(a, b, ap, bp);
  for (int kk = 1; kk < kTiles; ++kk) {
    ap += 32; bp += 32;
    __builtin_prefetch(bp + 32, 0, 3);   // global_prefetch_b8 on following B fragment
    FragU an, bn;
    frag_load(an, bn, ap, bp);
    acc = __builtin_amdgcn_wmma_f32_16x16x32_f16(false, a.v, false, b.v,
                                                 (short)0, acc, false, false);
    a = an; b = bn;
  }
  acc = __builtin_amdgcn_wmma_f32_16x16x32_f16(false, a.v, false, b.v,
                                               (short)0, acc, false, false);
  return acc;
}

__device__ __forceinline__ v8f zero8() { v8f z; for (int q=0;q<8;q++) z[q]=0.f; return z; }

// ---------------- K0: pack weights to padded f16 (optionally transposed) ----------
__global__ void k_pack(const float* src, _Float16* dst, int K, int N, int R, int C, int tr)
{
  int i = blockIdx.x*256 + threadIdx.x;
  if (i >= R*C) return;
  int r = i / C, c = i - r*C;
  float v = 0.f;
  if (tr) { if (r < N && c < K) v = src[c*N + r]; }    // dst[n][k] = W[k][n]
  else    { if (r < K && c < N) v = src[r*N + c]; }    // dst[k][n] padded
  dst[i] = (_Float16)v;
}

// ---------------- K1: per (frame,type) input stats; zero accumulators -------------
__global__ void k_stats(const float* sym, const int* types, float* stats, float* gacc, float* outE)
{
  const int f = blockIdx.x, tid = threadIdx.x;
  const float* sp = sym + (size_t)f*1024*128*4;
  const int* tp = types + f*1024;
  float ss[2]={0,0}, ss2[2]={0,0}, sx[2]={0,0}, sx2[2]={0,0}, cnt[2]={0,0};
  for (int n = tid; n < 1024; n += 256) {
    int t = tp[n];
    cnt[t] += 1.f;
    const float4* row = (const float4*)sp + n*128;
    float a0=0,a1=0,a2=0,a3=0;
    for (int s = 0; s < 128; ++s) {
      float4 v = row[s];
      a0 += v.x; a1 += v.x*v.x;
      a2 += v.y+v.z+v.w; a3 += v.y*v.y + v.z*v.z + v.w*v.w;
    }
    ss[t]+=a0; ss2[t]+=a1; sx[t]+=a2; sx2[t]+=a3;
  }
  __shared__ float red[256];
  float vals[10] = {ss[0],ss2[0],sx[0],sx2[0],cnt[0],ss[1],ss2[1],sx[1],sx2[1],cnt[1]};
  float out[10];
  for (int q=0;q<10;q++){
    red[tid]=vals[q]; __syncthreads();
    for (int s=128;s>0;s>>=1){ if(tid<s) red[tid]+=red[tid+s]; __syncthreads(); }
    out[q]=red[0]; __syncthreads();
  }
  if (tid==0){
    outE[f]=0.f;
    for(int t=0;t<2;t++){
      float S=out[t*5+0],S2=out[t*5+1],X=out[t*5+2],X2=out[t*5+3],C=out[t*5+4];
      float cf = fmaxf(C,1.f);
      float ns = cf*128.f, nx = ns*3.f;
      float sa=S/ns, xa=X/nx, sa2=S2/ns, xa2=X2/nx;
      float* st = stats + (f*2+t)*4;
      st[0]=sqrtf(fmaxf(sa2-sa*sa,0.f))+1e-8f;   // std_s
      st[1]=sqrtf(fmaxf(xa2-xa*xa,0.f))+1e-8f;   // std_x
      st[2]=sa; st[3]=cf;
      float* ga = gacc + (f*2+t)*4;
      ga[0]=ga[1]=ga[2]=ga[3]=0.f;
    }
  }
}

// ---------------- K2: embedding forward (fit MLP + RG + GRRG + stat atomics) ------
__global__ __launch_bounds__(256) void k_embed_fwd(
    const float* sym, const int* types,
    const float* fw0, const float* fb0, const float* fb1, const float* fb2,
    const _Float16* W1t, const _Float16* W2t,
    const float* stats, float* gacc, _Float16* grrg)
{
  extern __shared__ char smem[];
  float*    Rn  = (float*)(smem);                 // 128x4 f32
  _Float16* A1  = (_Float16*)(smem + 2048);       // 128x32 f16
  _Float16* A2  = (_Float16*)(smem + 10240);      // 128x64 f16
  float*    G   = (float*)(smem + 26624);         // 128x112 f32
  float*    RGs = (float*)(smem + 83968);         // 4x112 f32
  float*    red = (float*)(smem + 85760);         // 256 f32
  float*    b1p = (float*)(smem + 86784);         // 64 f32 (padded)
  float*    b2p = (float*)(smem + 87040);         // 112 f32 (padded)

  const int atom = blockIdx.x;
  const int f = atom >> 10;
  const int tid = threadIdx.x, lane = tid & 31, wave = tid >> 5;
  const int t = types[atom];
  const float* st = stats + (f*2+t)*4;
  const float stds = st[0], stdx = st[1], savg = st[2];
  const float* symp = sym + (size_t)atom*512;

  for (int i = tid; i < 512; i += 256) {
    int c = i & 3; float v = symp[i];
    Rn[i] = (c==0) ? (v - savg)/stds : v/stdx;
  }
  for (int i = tid; i < 64; i += 256)  b1p[i] = (i<50)  ? fb1[t*50+i]  : 0.f;
  for (int i = tid; i < 112; i += 256) b2p[i] = (i<100) ? fb2[t*100+i] : 0.f;
  __syncthreads();
  const float* w0 = fw0 + t*25;
  const float* b0 = fb0 + t*25;
  for (int i = tid; i < 128*32; i += 256) {
    int s = i>>5, j = i&31;
    A1[i] = (_Float16)((j<25) ? fast_tanh(Rn[s*4]*w0[j] + b0[j]) : 0.f);
  }
  __syncthreads();
  { const _Float16* W = W1t + t*64*32;
    for (int tile = wave; tile < 32; tile += 8) {
      int mt = tile>>2, nt = tile&3;
      v8f acc = wmma_tile(A1 + mt*16*32, 32, W + nt*16*32, 32, 1, lane, zero8());
      int m0 = lane&15, g = lane>>4, col = nt*16+m0;
      float bb = b1p[col];
      for (int r=0;r<8;r++) A2[(mt*16+r+8*g)*64 + col] = (_Float16)fast_tanh(acc[r]+bb);
    } }
  __syncthreads();
  { const _Float16* W = W2t + t*112*64;
    for (int tile = wave; tile < 56; tile += 8) {
      int mt = tile/7, nt = tile%7;
      v8f acc = wmma_tile(A2 + mt*16*64, 64, W + nt*16*64, 64, 2, lane, zero8());
      int m0 = lane&15, g = lane>>4, col = nt*16+m0;
      float bb = b2p[col];
      for (int r=0;r<8;r++) G[(mt*16+r+8*g)*112 + col] = fast_tanh(acc[r]+bb);
    } }
  __syncthreads();
  for (int i = tid; i < 448; i += 256) {
    int f4 = i/112, m = i - f4*112;
    float a = 0.f;
    for (int s=0;s<128;s++) a += Rn[s*4+f4]*G[s*112+m];
    RGs[i] = a;
  }
  __syncthreads();
  float lsum=0.f, lsq=0.f;
  for (int i = tid; i < 1600; i += 256) {
    int m = i >> 4, a = i & 15;
    float v = RGs[m]*RGs[a] + RGs[112+m]*RGs[112+a]
            + RGs[224+m]*RGs[224+a] + RGs[336+m]*RGs[336+a];
    grrg[(size_t)atom*1600 + i] = (_Float16)v;
    lsum += v; lsq += v*v;
  }
  red[tid]=lsum; __syncthreads();
  for (int s=128;s>0;s>>=1){ if(tid<s) red[tid]+=red[tid+s]; __syncthreads(); }
  if (tid==0) atomicAdd(&gacc[(f*2+t)*4+0], red[0]);
  __syncthreads();
  red[tid]=lsq; __syncthreads();
  for (int s=128;s>0;s>>=1){ if(tid<s) red[tid]+=red[tid+s]; __syncthreads(); }
  if (tid==0) atomicAdd(&gacc[(f*2+t)*4+1], red[0]);
}

// ---------------- K3: finalize GRRG stats ----------------------------------------
__global__ void k_gstats(const float* stats, const float* gacc, float* gst)
{
  int i = threadIdx.x;
  if (i < 8) {
    float cnt = stats[i*4+3];
    float denom = cnt * 1600.f;
    float mu  = gacc[i*4+0]/denom;
    float mu2 = gacc[i*4+1]/denom;
    float r = sqrtf(fmaxf(mu2 - mu*mu, 0.f));
    gst[i*4+0]=mu; gst[i*4+1]=r+1e-8f; gst[i*4+2]=r; gst[i*4+3]=0.f;
  }
}

// ---------------- K4: g-net forward (16 atoms per block, two type passes) ---------
__global__ __launch_bounds__(256) void k_gfwd(
    const int* types, const _Float16* grrg,
    const _Float16* G0t, const _Float16* G1t, const _Float16* G2t,
    const float* gb0, const float* gb1, const float* gb2,
    const float* gwo, const float* gbo, const float* gst,
    _Float16* h0o, _Float16* h1o, _Float16* h2o, float* outE)
{
  extern __shared__ char smem[];
  _Float16* D   = (_Float16*)smem;                // 16x1600 f16
  _Float16* hA  = (_Float16*)(smem + 51200);      // 16x256 f16
  _Float16* hB  = (_Float16*)(smem + 59392);      // 16x256 f16
  float*    red = (float*)(smem + 67584);         // 256 f32
  float*    bia = (float*)(smem + 68608);         // 240 f32
  float*    gwoL= (float*)(smem + 69632);         // 240 f32
  __shared__ int ltype[16];
  const int blk = blockIdx.x;
  const int f = blk >> 6;
  const int base = f*1024 + (blk & 63)*16;
  const int tid = threadIdx.x, lane = tid&31, wave = tid>>5;
  if (tid < 16) ltype[tid] = types[base + tid];
  for (int t = 0; t < 2; ++t) {
    __syncthreads();
    const float mu = gst[(f*2+t)*4], inv = 1.f/gst[(f*2+t)*4+1];
    for (int i = tid; i < 16*1600; i += 256) {
      int a = i/1600, k = i - a*1600;
      D[i] = (_Float16)(((float)grrg[(size_t)(base+a)*1600 + k] - mu)*inv);
    }
    for (int i = tid; i < 16*256; i += 256) hA[i] = (_Float16)0.f;
    for (int i = tid; i < 240; i += 256) { bia[i] = gb0[t*240+i]; gwoL[i] = gwo[t*240+i]; }
    __syncthreads();
    { const _Float16* B = G0t + t*240*1600;
      for (int nt = wave; nt < 15; nt += 8) {
        v8f acc = wmma_tile(D, 1600, B + nt*16*1600, 1600, 50, lane, zero8());
        int m0=lane&15, g=lane>>4, col=nt*16+m0;
        float bv = bia[col];
        for (int r=0;r<8;r++){
          int a = r + 8*g;
          float h = fast_tanh(acc[r]+bv);
          hA[a*256+col] = (_Float16)h;
          if (ltype[a]==t) h0o[(size_t)(base+a)*240 + col] = (_Float16)h;
        } } }
    __syncthreads();
    for (int i = tid; i < 16*256; i += 256) hB[i] = (_Float16)0.f;
    for (int i = tid; i < 240; i += 256) bia[i] = gb1[t*240+i];
    __syncthreads();
    { const _Float16* B = G1t + t*240*256;
      for (int nt = wave; nt < 15; nt += 8) {
        v8f acc = wmma_tile(hA, 256, B + nt*16*256, 256, 8, lane, zero8());
        int m0=lane&15, g=lane>>4, col=nt*16+m0;
        float bv = bia[col];
        for (int r=0;r<8;r++){
          int a = r + 8*g;
          float h = fast_tanh(acc[r]+bv);
          hB[a*256+col] = (_Float16)h;
          if (ltype[a]==t) h1o[(size_t)(base+a)*240 + col] = (_Float16)h;
        } } }
    __syncthreads();
    for (int i = tid; i < 16*256; i += 256) hA[i] = (_Float16)0.f;
    for (int i = tid; i < 240; i += 256) bia[i] = gb2[t*240+i];
    __syncthreads();
    { const _Float16* B = G2t + t*240*256;
      for (int nt = wave; nt < 15; nt += 8) {
        v8f acc = wmma_tile(hB, 256, B + nt*16*256, 256, 8, lane, zero8());
        int m0=lane&15, g=lane>>4, col=nt*16+m0;
        float bv = bia[col];
        for (int r=0;r<8;r++){
          int a = r + 8*g;
          float h = fast_tanh(acc[r]+bv);
          hA[a*256+col] = (_Float16)h;
          if (ltype[a]==t) h2o[(size_t)(base+a)*240 + col] = (_Float16)h;
        } } }
    __syncthreads();
    { // e = h2 . gwo  (16 threads per atom)
      int a = tid >> 4, j = tid & 15;
      float p = 0.f;
      for (int o = j; o < 240; o += 16) p += (float)hA[a*256+o]*gwoL[o];
      red[tid] = p; __syncthreads();
      for (int s=8;s>0;s>>=1){ if (j<s) red[tid]+=red[tid+s]; __syncthreads(); }
      if (j==0 && ltype[a]==t) atomicAdd(&outE[f], red[a*16]+gbo[t]);
    }
  }
}

// ---------------- K5: g-net backward to gD; accumulate S1,S2 ----------------------
__global__ __launch_bounds__(256) void k_gbwd(
    const int* types, const _Float16* grrg,
    const _Float16* G0b, const _Float16* G1b, const _Float16* G2b,
    const float* gwo,
    const _Float16* h0o, const _Float16* h1o, const _Float16* h2o,
    const float* gst, float* gacc, float* gD)
{
  extern __shared__ char smem[];
  _Float16* zA = (_Float16*)smem;            // 16x256
  _Float16* zB = (_Float16*)(smem + 8192);   // 16x256
  float* red   = (float*)(smem + 16384);     // 256 f32
  float* gwoL  = (float*)(smem + 17408);     // 240 f32
  __shared__ int ltype[16];
  const int blk = blockIdx.x;
  const int f = blk >> 6;
  const int base = f*1024 + (blk & 63)*16;
  const int tid = threadIdx.x, lane = tid&31, wave = tid>>5;
  if (tid < 16) ltype[tid] = types[base + tid];
  for (int t = 0; t < 2; ++t) {
    __syncthreads();
    const float mu = gst[(f*2+t)*4], sdv = gst[(f*2+t)*4+1];
    for (int i = tid; i < 240; i += 256) gwoL[i] = gwo[t*240+i];
    __syncthreads();
    for (int i = tid; i < 16*256; i += 256) {
      int a = i>>8, o = i&255;
      float v = 0.f;
      if (o < 240 && ltype[a]==t) {
        float h2 = (float)h2o[(size_t)(base+a)*240+o];
        v = gwoL[o]*(1.f-h2*h2);
      }
      zA[i] = (_Float16)v;
    }
    for (int i = tid; i < 16*256; i += 256) zB[i] = (_Float16)0.f;
    __syncthreads();
    { const _Float16* B = G2b + t*240*256;
      for (int nt = wave; nt < 15; nt += 8) {
        v8f acc = wmma_tile(zA, 256, B + nt*16*256, 256, 8, lane, zero8());
        int m0=lane&15, g=lane>>4, col=nt*16+m0;
        for (int r=0;r<8;r++){
          int a = r+8*g;
          float h1 = (ltype[a]==t) ? (float)h1o[(size_t)(base+a)*240+col] : 0.f;
          zB[a*256+col] = (_Float16)(acc[r]*(1.f-h1*h1));
        } } }
    __syncthreads();
    for (int i = tid; i < 16*256; i += 256) zA[i] = (_Float16)0.f;
    __syncthreads();
    { const _Float16* B = G1b + t*240*256;
      for (int nt = wave; nt < 15; nt += 8) {
        v8f acc = wmma_tile(zB, 256, B + nt*16*256, 256, 8, lane, zero8());
        int m0=lane&15, g=lane>>4, col=nt*16+m0;
        for (int r=0;r<8;r++){
          int a = r+8*g;
          float h0 = (ltype[a]==t) ? (float)h0o[(size_t)(base+a)*240+col] : 0.f;
          zA[a*256+col] = (_Float16)(acc[r]*(1.f-h0*h0));
        } } }
    __syncthreads();
    float s1=0.f, s2=0.f;
    { const _Float16* B = G0b + t*1600*256;
      for (int nt = wave; nt < 100; nt += 8) {
        v8f acc = wmma_tile(zA, 256, B + nt*16*256, 256, 8, lane, zero8());
        int m0=lane&15, g=lane>>4, col=nt*16+m0;
        for (int r=0;r<8;r++){
          int a = r+8*g;
          if (ltype[a]==t) {
            size_t gi = (size_t)(base+a)*1600 + col;
            float gd = acc[r];
            gD[gi] = gd;
            float d = ((float)grrg[gi]-mu)/sdv;
            s1 += gd; s2 += gd*d;
          } } } }
    red[tid]=s1; __syncthreads();
    for (int ss=128;ss>0;ss>>=1){ if(tid<ss) red[tid]+=red[tid+ss]; __syncthreads(); }
    if (tid==0) atomicAdd(&gacc[(f*2+t)*4+2], red[0]);
    __syncthreads();
    red[tid]=s2; __syncthreads();
    for (int ss=128;ss>0;ss>>=1){ if(tid<ss) red[tid]+=red[tid+ss]; __syncthreads(); }
    if (tid==0) atomicAdd(&gacc[(f*2+t)*4+3], red[0]);
  }
}

// ---------------- K6: embedding backward + forces ---------------------------------
__global__ __launch_bounds__(256) void k_embed_bwd(
    const float* sym, const float* dxg, const float* dyg, const float* dzg,
    const int* types,
    const float* fw0, const float* fb0, const float* fb1, const float* fb2,
    const _Float16* W1t, const _Float16* W2t,
    const _Float16* W2b, const _Float16* W1b,
    const float* stats, const float* gst, const float* gacc,
    const _Float16* grrg, const float* gD, float* outF)
{
  extern __shared__ char smem[];
  float*    Rn  = (float*)(smem);                 // 2048
  _Float16* A1  = (_Float16*)(smem + 2048);       // 8192
  _Float16* A2  = (_Float16*)(smem + 10240);      // 16384
  float*    G   = (float*)(smem + 26624);         // 57344
  float*    RGs = (float*)(smem + 83968);         // 1792
  float*    dRG = (float*)(smem + 85760);         // 1792
  float*    dGR = (float*)(smem + 87552);         // 6400
  _Float16* Z2  = (_Float16*)(smem + 93952);      // 32768 (128x128)
  _Float16* Z1  = (_Float16*)(smem + 126720);     // 16384 (128x64)
  float*    Z0  = (float*)(smem + 143104);        // 16384 (128x32 f32)
  float*    red = (float*)(smem + 159488);        // 1024
  float*    b1p = (float*)(smem + 160512);        // 64 f32
  float*    b2p = (float*)(smem + 160768);        // 112 f32

  const int atom = blockIdx.x;
  const int f = atom >> 10;
  const int tid = threadIdx.x, lane = tid&31, wave = tid>>5;
  const int t = types[atom];
  const float* st = stats + (f*2+t)*4;
  const float stds = st[0], stdx = st[1], savg = st[2], cnt = st[3];
  const float mu = gst[(f*2+t)*4], sdv = gst[(f*2+t)*4+1], rr = gst[(f*2+t)*4+2];
  const float S1 = gacc[(f*2+t)*4+2], S2 = gacc[(f*2+t)*4+3];
  const float denom = cnt*1600.f;
  const float c1 = S1/(sdv*denom);
  const float c2 = S2/(fmaxf(rr,1e-30f)*denom);
  const float* symp = sym + (size_t)atom*512;

  for (int i = tid; i < 512; i += 256) {
    int c = i & 3; float v = symp[i];
    Rn[i] = (c==0) ? (v - savg)/stds : v/stdx;
  }
  for (int i = tid; i < 64; i += 256)  b1p[i] = (i<50)  ? fb1[t*50+i]  : 0.f;
  for (int i = tid; i < 112; i += 256) b2p[i] = (i<100) ? fb2[t*100+i] : 0.f;
  __syncthreads();
  const float* w0 = fw0 + t*25;
  const float* b0 = fb0 + t*25;
  for (int i = tid; i < 128*32; i += 256) {
    int s = i>>5, j = i&31;
    A1[i] = (_Float16)((j<25) ? fast_tanh(Rn[s*4]*w0[j] + b0[j]) : 0.f);
  }
  __syncthreads();
  { const _Float16* W = W1t + t*64*32;
    for (int tile = wave; tile < 32; tile += 8) {
      int mt = tile>>2, nt = tile&3;
      v8f acc = wmma_tile(A1 + mt*16*32, 32, W + nt*16*32, 32, 1, lane, zero8());
      int m0=lane&15, g=lane>>4, col=nt*16+m0;
      float bb = b1p[col];
      for (int r=0;r<8;r++) A2[(mt*16+r+8*g)*64+col] = (_Float16)fast_tanh(acc[r]+bb);
    } }
  __syncthreads();
  { const _Float16* W = W2t + t*112*64;
    for (int tile = wave; tile < 56; tile += 8) {
      int mt = tile/7, nt = tile%7;
      v8f acc = wmma_tile(A2 + mt*16*64, 64, W + nt*16*64, 64, 2, lane, zero8());
      int m0=lane&15, g=lane>>4, col=nt*16+m0;
      float bb = b2p[col];
      for (int r=0;r<8;r++) G[(mt*16+r+8*g)*112+col] = fast_tanh(acc[r]+bb);
    } }
  __syncthreads();
  for (int i = tid; i < 448; i += 256) {
    int f4 = i/112, m = i - f4*112;
    float a = 0.f;
    for (int s=0;s<128;s++) a += Rn[s*4+f4]*G[s*112+m];
    RGs[i] = a;
  }
  const float* gDp = gD + (size_t)atom*1600;
  const _Float16* grp = grrg + (size_t)atom*1600;
  for (int i = tid; i < 1600; i += 256) {
    float gd = gDp[i];
    float d = ((float)grp[i] - mu)/sdv;
    dGR[i] = gd/sdv - c1 - d*c2;
  }
  __syncthreads();
  for (int i = tid; i < 448; i += 256) {
    int f4 = i/112, k = i - f4*112;
    float a = 0.f;
    if (k < 100) {
      for (int aa=0; aa<16; ++aa) a += dGR[k*16+aa]*RGs[f4*112+aa];
      if (k < 16) for (int m=0;m<100;m++) a += dGR[m*16+k]*RGs[f4*112+m];
    }
    dRG[i] = a;
  }
  __syncthreads();
  for (int i = tid; i < 128*128; i += 256) {
    int s = i>>7, m = i&127;
    float v = 0.f;
    if (m < 112) {
      float gv = G[s*112+m];
      float dg = Rn[s*4+0]*dRG[m] + Rn[s*4+1]*dRG[112+m]
               + Rn[s*4+2]*dRG[224+m] + Rn[s*4+3]*dRG[336+m];
      v = dg*(1.f-gv*gv);
    }
    Z2[i] = (_Float16)v;
  }
  __syncthreads();
  { const _Float16* W = W2b + t*64*128;
    for (int tile = wave; tile < 32; tile += 8) {
      int mt = tile>>2, nt = tile&3;
      v8f acc = wmma_tile(Z2 + mt*16*128, 128, W + nt*16*128, 128, 4, lane, zero8());
      int m0=lane&15, g=lane>>4, col=nt*16+m0;
      for (int r=0;r<8;r++){
        int row = mt*16+r+8*g;
        float a2 = (float)A2[row*64+col];
        Z1[row*64+col] = (_Float16)(acc[r]*(1.f-a2*a2));
      } } }
  __syncthreads();
  { const _Float16* W = W1b + t*32*64;
    for (int tile = wave; tile < 16; tile += 8) {
      int mt = tile>>1, nt = tile&1;
      v8f acc = wmma_tile(Z1 + mt*16*64, 64, W + nt*16*64, 64, 2, lane, zero8());
      int m0=lane&15, g=lane>>4, col=nt*16+m0;
      for (int r=0;r<8;r++){
        int row = mt*16+r+8*g;
        float a1 = (float)A1[row*32+col];
        Z0[row*32+col] = acc[r]*(1.f-a1*a1);
      } } }
  __syncthreads();
  const float* dxp = dxg + (size_t)atom*512;
  const float* dyp = dyg + (size_t)atom*512;
  const float* dzp = dzg + (size_t)atom*512;
  float fx=0.f, fy=0.f, fz=0.f;
  for (int i = tid; i < 512; i += 256) {
    int s = i>>2, c = i&3;
    float dr = 0.f;
    for (int m=0;m<100;m++) dr += dRG[c*112+m]*G[s*112+m];
    if (c==0) {
      float ds = 0.f;
      for (int j=0;j<25;j++) ds += Z0[s*32+j]*w0[j];
      dr += ds;
    }
    float drn = dr * ((c==0)? (1.f/stds) : (1.f/stdx));
    fx += drn*dxp[i]; fy += drn*dyp[i]; fz += drn*dzp[i];
  }
  red[tid]=fx; __syncthreads();
  for (int s2=128;s2>0;s2>>=1){ if(tid<s2) red[tid]+=red[tid+s2]; __syncthreads(); }
  float Fx = red[0]; __syncthreads();
  red[tid]=fy; __syncthreads();
  for (int s2=128;s2>0;s2>>=1){ if(tid<s2) red[tid]+=red[tid+s2]; __syncthreads(); }
  float Fy = red[0]; __syncthreads();
  red[tid]=fz; __syncthreads();
  for (int s2=128;s2>0;s2>>=1){ if(tid<s2) red[tid]+=red[tid+s2]; __syncthreads(); }
  if (tid==0){
    outF[atom*3+0] = -Fx;
    outF[atom*3+1] = -Fy;
    outF[atom*3+2] = -red[0];
  }
}

// ---------------- host ------------------------------------------------------------
extern "C" void kernel_launch(void* const* d_in, const int* in_sizes, int n_in,
                              void* d_out, int out_size, void* d_ws, size_t ws_size,
                              hipStream_t stream)
{
  (void)in_sizes; (void)n_in; (void)out_size; (void)ws_size;
  const float* sym = (const float*)d_in[0];
  const float* dxg = (const float*)d_in[1];
  const float* dyg = (const float*)d_in[2];
  const float* dzg = (const float*)d_in[3];
  const int*   types = (const int*)d_in[4];
  const float* fw0 = (const float*)d_in[5];
  const float* fb0 = (const float*)d_in[6];
  const float* fw1 = (const float*)d_in[7];
  const float* fb1 = (const float*)d_in[8];
  const float* fw2 = (const float*)d_in[9];
  const float* fb2 = (const float*)d_in[10];
  const float* gw0 = (const float*)d_in[11];
  const float* gb0 = (const float*)d_in[12];
  const float* gw1 = (const float*)d_in[13];
  const float* gb1 = (const float*)d_in[14];
  const float* gw2 = (const float*)d_in[15];
  const float* gb2 = (const float*)d_in[16];
  const float* gwo = (const float*)d_in[17];
  const float* gbo = (const float*)d_in[18];
  float* outE = (float*)d_out;
  float* outF = (float*)d_out + 4;

  char* ws = (char*)d_ws;
  size_t off = 0;
  auto take = [&](size_t bytes) -> char* {
    off = (off + 255) & ~(size_t)255;
    char* p = ws + off; off += bytes; return p;
  };
  float* stats = (float*)take(32*4);
  float* gacc  = (float*)take(32*4);
  float* gst   = (float*)take(32*4);
  _Float16* W1t = (_Float16*)take(2*2048*2);
  _Float16* W1b = (_Float16*)take(2*2048*2);
  _Float16* W2t = (_Float16*)take(2*7168*2);
  _Float16* W2b = (_Float16*)take(2*8192*2);
  _Float16* G0t = (_Float16*)take((size_t)2*384000*2);
  _Float16* G0b = (_Float16*)take((size_t)2*409600*2);
  _Float16* G1t = (_Float16*)take((size_t)2*61440*2);
  _Float16* G1b = (_Float16*)take((size_t)2*61440*2);
  _Float16* G2t = (_Float16*)take((size_t)2*61440*2);
  _Float16* G2b = (_Float16*)take((size_t)2*61440*2);
  _Float16* GR  = (_Float16*)take((size_t)4096*1600*2);
  _Float16* H0  = (_Float16*)take((size_t)4096*240*2);
  _Float16* H1  = (_Float16*)take((size_t)4096*240*2);
  _Float16* H2  = (_Float16*)take((size_t)4096*240*2);
  float*    GD  = (float*)take((size_t)4096*1600*4);

  auto pk = [&](const float* src, _Float16* dst, int K, int N, int R, int C, int tr){
    int total = R*C;
    k_pack<<<dim3((total+255)/256), dim3(256), 0, stream>>>(src, dst, K, N, R, C, tr);
  };
  for (int t=0;t<2;t++){
    pk(fw1 + t*1250,   W1t + t*2048,   25, 50,  64, 32,  1);
    pk(fw1 + t*1250,   W1b + t*2048,   25, 50,  32, 64,  0);
    pk(fw2 + t*5000,   W2t + t*7168,   50, 100, 112, 64, 1);
    pk(fw2 + t*5000,   W2b + t*8192,   50, 100, 64, 128, 0);
    pk(gw0 + t*384000, G0t + t*384000, 1600, 240, 240, 1600, 1);
    pk(gw0 + t*384000, G0b + t*409600, 1600, 240, 1600, 256, 0);
    pk(gw1 + t*57600,  G1t + t*61440,  240, 240, 240, 256, 1);
    pk(gw1 + t*57600,  G1b + t*61440,  240, 240, 240, 256, 0);
    pk(gw2 + t*57600,  G2t + t*61440,  240, 240, 240, 256, 1);
    pk(gw2 + t*57600,  G2b + t*61440,  240, 240, 240, 256, 0);
  }
  k_stats<<<dim3(4), dim3(256), 0, stream>>>(sym, types, stats, gacc, outE);
  k_embed_fwd<<<dim3(4096), dim3(256), 87488, stream>>>(
      sym, types, fw0, fb0, fb1, fb2, W1t, W2t, stats, gacc, GR);
  k_gstats<<<dim3(1), dim3(32), 0, stream>>>(stats, gacc, gst);
  k_gfwd<<<dim3(256), dim3(256), 70592, stream>>>(
      types, GR, G0t, G1t, G2t, gb0, gb1, gb2, gwo, gbo, gst, H0, H1, H2, outE);
  k_gbwd<<<dim3(256), dim3(256), 18368, stream>>>(
      types, GR, G0b, G1b, G2b, gwo, H0, H1, H2, gst, gacc, GD);
  k_embed_bwd<<<dim3(4096), dim3(256), 161216, stream>>>(
      sym, dxg, dyg, dzg, types, fw0, fb0, fb1, fb2, W1t, W2t, W2b, W1b,
      stats, gst, gacc, GR, GD, outF);
}
